// ArchSampler_45251775431136
// MI455X (gfx1250) — matrix operand order repaired and checked by reading, not compiled, not run on hardware
//
#include <hip/hip_runtime.h>

// ---------------------------------------------------------------------------
// MI455X (gfx1250) ArchSampler: 24 sequential steps of 2-layer LSTM(H=2048)
// + Gumbel-max sampling.  Bandwidth-bound: w_ih/w_hh = 268MB fp32 read 24x.
// Strategy: pack weights to bf16 (134MB, fits 192MB L2) in the CDNA5 WMMA
// A-operand register layout once per launch, then drive every GEMV through
// V_WMMA_F32_16X16X32_BF16 with the vector broadcast across B's 16 columns.
// ---------------------------------------------------------------------------

#define H        2048
#define G4H      8192              // 4*H gate rows
#define TILE_K   32
#define KTILES   (H / TILE_K)      // 64
#define MTILES   (G4H / 16)        // 512
#define TILE_ELEMS 512             // 32 lanes * 16 bf16 per WMMA A tile
#define LAYER_ELEMS (G4H * H)      // 16,777,216 elems per (tensor,layer)

typedef __attribute__((ext_vector_type(16))) __bf16 v16bf;
typedef __attribute__((ext_vector_type(8)))  float  v8f;

struct alignas(16) U4 { unsigned int a, b, c, d; };
struct alignas(16) F4 { float x, y, z, w; };
union BF16x16 { U4 u[2]; v16bf v; unsigned short s[16]; };

static __device__ __forceinline__ unsigned short bf16rne(float f) {
    unsigned int u = __float_as_uint(f);
    unsigned int r = u + 0x7FFFu + ((u >> 16) & 1u);   // round-to-nearest-even
    return (unsigned short)(r >> 16);
}

static __device__ __forceinline__ v8f wmma_bf16(v16bf a, v16bf b, v8f c) {
    // D = A(16x32 bf16) * B(32x16 bf16) + C(16x16 f32)
    return __builtin_amdgcn_wmma_f32_16x16x32_bf16(
        false, a, false, b, (short)0, c, false, false);
}

// A tile pre-swizzled in memory: lane*16 contiguous bf16 == the lane's A regs.
static __device__ __forceinline__ v16bf load_A_packed(const unsigned short* t, int lane) {
    BF16x16 r;
    const U4* p = (const U4*)(t + lane * 16);
    r.u[0] = p[0]; r.u[1] = p[1];
    return r.v;
}

// Fallback: gather 16 f32 weights per lane (rows stride H) and convert.
static __device__ __forceinline__ v16bf load_A_raw(const float* rowp, int k0) {
    F4 a0 = *(const F4*)(rowp + k0);
    F4 a1 = *(const F4*)(rowp + k0 + 4);
    F4 a2 = *(const F4*)(rowp + k0 + 16);
    F4 a3 = *(const F4*)(rowp + k0 + 20);
    BF16x16 r;
    r.s[0]  = bf16rne(a0.x); r.s[1]  = bf16rne(a0.y); r.s[2]  = bf16rne(a0.z); r.s[3]  = bf16rne(a0.w);
    r.s[4]  = bf16rne(a1.x); r.s[5]  = bf16rne(a1.y); r.s[6]  = bf16rne(a1.z); r.s[7]  = bf16rne(a1.w);
    r.s[8]  = bf16rne(a2.x); r.s[9]  = bf16rne(a2.y); r.s[10] = bf16rne(a2.z); r.s[11] = bf16rne(a2.w);
    r.s[12] = bf16rne(a3.x); r.s[13] = bf16rne(a3.y); r.s[14] = bf16rne(a3.z); r.s[15] = bf16rne(a3.w);
    return r.v;
}

// B operand: vector value broadcast to all 16 columns.  Lane's B regs are
// 8 bf16 at K offset (kt*32 + (lane/16)*8) then 8 bf16 at that offset + 16.
static __device__ __forceinline__ v16bf load_B(const unsigned short* sh, int off16) {
    BF16x16 r;
    const U4* p = (const U4*)(sh + off16);
    r.u[0] = p[0]; r.u[1] = p[2];
    return r.v;
}

// ---------------------------------------------------------------------------
// One-time (per launch) f32 -> bf16 pack into WMMA A-tile layout.
// Tile (mt,kt): elem (lane,j) = W[mt*16 + lane%16, kt*32 + (j/8)*16 + (lane/16)*8 + j%8]
// ---------------------------------------------------------------------------
__global__ void pack_weights(const float* __restrict__ w_ih,
                             const float* __restrict__ w_hh,
                             unsigned short* __restrict__ p_ih,
                             unsigned short* __restrict__ p_hh) {
    const size_t total = 4ull * LAYER_ELEMS;   // {ih,hh} x {layer0,layer1}
    for (size_t g = (size_t)blockIdx.x * blockDim.x + threadIdx.x;
         g < total; g += (size_t)gridDim.x * blockDim.x) {
        unsigned int t = (unsigned int)(g / LAYER_ELEMS);
        unsigned int e = (unsigned int)(g % LAYER_ELEMS);
        unsigned int mt    = e >> 15;
        unsigned int rem   = e & 32767u;
        unsigned int kt    = rem >> 9;
        unsigned int r2    = rem & 511u;
        unsigned int lane  = r2 >> 4;
        unsigned int j     = r2 & 15u;
        unsigned int m     = mt * 16 + (lane & 15u);
        unsigned int khalf = lane >> 4;
        unsigned int k     = kt * 32 + ((j >> 3) << 4) + khalf * 8 + (j & 7u);
        const float* src = ((t < 2) ? w_ih : w_hh) + (size_t)(t & 1u) * LAYER_ELEMS;
        unsigned short* dst = ((t < 2) ? p_ih : p_hh) + (size_t)(t & 1u) * LAYER_ELEMS + e;
        *dst = bf16rne(src[(size_t)m * H + k]);
    }
}

__global__ void init_state(float* h, float* c, float* xacc) {
    int i = blockIdx.x * blockDim.x + threadIdx.x;
    if (i < 2 * H) { h[i] = 0.0f; c[i] = 0.0f; }
    if (i == 0) *xacc = 0.0f;
}

// ---------------------------------------------------------------------------
// gates[8192] = w_ih @ x + w_hh @ h   (bf16 WMMA, f32 accumulate)
// One wave per 16-row tile; block = 128 threads (4 waves); grid = 128.
// ---------------------------------------------------------------------------
template <bool PACKED>
__global__ void gemv_gates(float* __restrict__ gates,
                           const unsigned short* __restrict__ pA_ih,
                           const float* __restrict__ rA_ih,
                           const unsigned short* __restrict__ pA_hh,
                           const float* __restrict__ rA_hh,
                           const float* __restrict__ x,
                           const float* __restrict__ h) {
    __shared__ __align__(16) unsigned short shx[H];
    __shared__ __align__(16) unsigned short shh[H];

    const int tid = threadIdx.x;
    for (int i = tid; i < H; i += blockDim.x) {
        shx[i] = bf16rne(x[i]);
        shh[i] = bf16rne(h[i]);
    }
    __syncthreads();

    const int lane  = tid & 31;
    const int wave  = tid >> 5;
    const int mt    = blockIdx.x * (blockDim.x >> 5) + wave;   // 0..511
    const int mrow  = lane & 15;
    const int khalf = lane >> 4;

    v8f acc0 = {0.f, 0.f, 0.f, 0.f, 0.f, 0.f, 0.f, 0.f};
    v8f acc1 = {0.f, 0.f, 0.f, 0.f, 0.f, 0.f, 0.f, 0.f};

    // ---- pass 1: w_ih @ x ----
    {
        const unsigned short* pA = PACKED ? (pA_ih + (size_t)mt * (KTILES * TILE_ELEMS)) : nullptr;
        const float* rowp = PACKED ? nullptr : (rA_ih + (size_t)(mt * 16 + mrow) * H);
        #pragma unroll 4
        for (int kt = 0; kt < KTILES; kt += 2) {
            if (PACKED)
                __builtin_prefetch(pA + (kt + 2) * TILE_ELEMS + lane * 16, 0, 0);
            v16bf a0 = PACKED ? load_A_packed(pA + kt * TILE_ELEMS, lane)
                              : load_A_raw(rowp, kt * TILE_K + khalf * 8);
            v16bf b0 = load_B(shx, kt * TILE_K + khalf * 8);
            acc0 = wmma_bf16(a0, b0, acc0);
            v16bf a1 = PACKED ? load_A_packed(pA + (kt + 1) * TILE_ELEMS, lane)
                              : load_A_raw(rowp, (kt + 1) * TILE_K + khalf * 8);
            v16bf b1 = load_B(shx, (kt + 1) * TILE_K + khalf * 8);
            acc1 = wmma_bf16(a1, b1, acc1);
        }
    }
    // ---- pass 2: w_hh @ h ----
    {
        const unsigned short* pA = PACKED ? (pA_hh + (size_t)mt * (KTILES * TILE_ELEMS)) : nullptr;
        const float* rowp = PACKED ? nullptr : (rA_hh + (size_t)(mt * 16 + mrow) * H);
        #pragma unroll 4
        for (int kt = 0; kt < KTILES; kt += 2) {
            if (PACKED)
                __builtin_prefetch(pA + (kt + 2) * TILE_ELEMS + lane * 16, 0, 0);
            v16bf a0 = PACKED ? load_A_packed(pA + kt * TILE_ELEMS, lane)
                              : load_A_raw(rowp, kt * TILE_K + khalf * 8);
            v16bf b0 = load_B(shh, kt * TILE_K + khalf * 8);
            acc0 = wmma_bf16(a0, b0, acc0);
            v16bf a1 = PACKED ? load_A_packed(pA + (kt + 1) * TILE_ELEMS, lane)
                              : load_A_raw(rowp, (kt + 1) * TILE_K + khalf * 8);
            v16bf b1 = load_B(shh, (kt + 1) * TILE_K + khalf * 8);
            acc1 = wmma_bf16(a1, b1, acc1);
        }
    }

    v8f acc = acc0 + acc1;
    // D layout (16x16 f32): lane L, vgpr v -> M = v + 8*(L/16), N = L%16.
    // All N columns identical (B broadcast); lanes with N==0 store.
    if (mrow == 0) {
        float* op = gates + mt * 16 + khalf * 8;
        F4 lo = {acc[0], acc[1], acc[2], acc[3]};
        F4 hi = {acc[4], acc[5], acc[6], acc[7]};
        *(F4*)(op)     = lo;
        *(F4*)(op + 4) = hi;
    }
}

// c2 = sig(f)*c + sig(i)*tanh(g) ; h2 = sig(o)*tanh(c2)   (gate order i,f,g,o)
__global__ void cell_update(const float* __restrict__ gates,
                            float* __restrict__ h, float* __restrict__ c) {
    int idx = blockIdx.x * blockDim.x + threadIdx.x;   // 0..2047
    float gi = gates[idx];
    float gf = gates[idx + H];
    float gg = gates[idx + 2 * H];
    float go = gates[idx + 3 * H];
    float si = 1.0f / (1.0f + expf(-gi));
    float sf = 1.0f / (1.0f + expf(-gf));
    float so = 1.0f / (1.0f + expf(-go));
    float c2 = sf * c[idx] + si * tanhf(gg);
    c[idx] = c2;
    h[idx] = so * tanhf(c2);
}

// logits = h1 @ w_soft (2048x8) -> log_softmax -> Gumbel-max -> xent; pick
// next input embedding.  Single workgroup (serial tail is 8-wide, trivial).
__global__ void sample_step(const float* __restrict__ h1,
                            const float* __restrict__ w_soft,
                            const float* __restrict__ w_emb,
                            const float* __restrict__ u,
                            int s,
                            float* __restrict__ x_next,
                            float* __restrict__ xacc,
                            float* __restrict__ out) {
    __shared__ float sred[256 * 8];
    __shared__ float slog[8];
    __shared__ int   sidx;
    const int tid = threadIdx.x;

    float acc[8] = {0.f, 0.f, 0.f, 0.f, 0.f, 0.f, 0.f, 0.f};
    for (int k = tid; k < H; k += 256) {
        float hv = h1[k];
        #pragma unroll
        for (int t = 0; t < 8; ++t) acc[t] += hv * w_soft[k * 8 + t];
    }
    #pragma unroll
    for (int t = 0; t < 8; ++t) sred[tid * 8 + t] = acc[t];
    __syncthreads();

    if (tid < 8) {
        float ssum = 0.f;
        for (int i = 0; i < 256; ++i) ssum += sred[i * 8 + tid];
        slog[tid] = ssum;
    }
    __syncthreads();

    if (tid == 0) {
        float m = -1e30f;
        for (int t = 0; t < 8; ++t) m = fmaxf(m, slog[t]);
        float se = 0.f;
        for (int t = 0; t < 8; ++t) se += expf(slog[t] - m);
        float lse = m + logf(se);
        int best = 0; float bs = -1e30f; float blp = 0.f;
        for (int t = 0; t < 8; ++t) {
            float lp = slog[t] - lse;
            float gb = -logf(-logf(u[s * 8 + t]));
            float sc = lp + gb;
            if (sc > bs) { bs = sc; best = t; blp = lp; }
        }
        float xent = -blp;
        float a = ((s == 0) ? 0.f : *xacc) + xent;
        *xacc = a;
        out[s] = (float)best;
        if (s == 23) out[24] = a / 24.0f;
        sidx = best;
    }
    __syncthreads();

    const int idx = sidx;
    const float* emb = w_emb + ((size_t)s * 8 + idx) * H;
    for (int k = tid; k < H; k += 256) x_next[k] = emb[k];
}

// ---------------------------------------------------------------------------
extern "C" void kernel_launch(void* const* d_in, const int* in_sizes, int n_in,
                              void* d_out, int out_size, void* d_ws, size_t ws_size,
                              hipStream_t stream) {
    (void)in_sizes; (void)n_in; (void)out_size;
    const float* g_emb  = (const float*)d_in[0];
    const float* w_emb  = (const float*)d_in[1];
    const float* w_soft = (const float*)d_in[2];
    const float* w_ih   = (const float*)d_in[3];
    const float* w_hh   = (const float*)d_in[4];
    const float* u      = (const float*)d_in[5];
    float* out = (float*)d_out;

    const size_t packBytesPer = 2ull * LAYER_ELEMS * sizeof(unsigned short); // 67.1MB
    const size_t packBytes    = 2ull * packBytesPer;                         // 134.2MB
    const bool   packed       = ws_size >= packBytes + (1u << 20);

    char* base = (char*)d_ws;
    unsigned short* p_ih = (unsigned short*)base;
    unsigned short* p_hh = (unsigned short*)(base + packBytesPer);
    float* fbase = (float*)(base + (packed ? packBytes : 0));
    float* xv    = fbase;            // H
    float* hbuf  = xv + H;           // 2*H
    float* cbuf  = hbuf + 2 * H;     // 2*H
    float* gates = cbuf + 2 * H;     // 4*H
    float* xacc  = gates + 4 * H;    // 1

    init_state<<<16, 256, 0, stream>>>(hbuf, cbuf, xacc);
    if (packed)
        pack_weights<<<8192, 256, 0, stream>>>(w_ih, w_hh, p_ih, p_hh);

    for (int s = 0; s < 24; ++s) {
        for (int l = 0; l < 2; ++l) {
            const float* xin = (l == 0) ? ((s == 0) ? g_emb : xv) : hbuf;
            const float* hin = hbuf + l * H;
            if (packed) {
                gemv_gates<true><<<MTILES / 4, 128, 0, stream>>>(
                    gates,
                    p_ih + (size_t)l * LAYER_ELEMS, nullptr,
                    p_hh + (size_t)l * LAYER_ELEMS, nullptr,
                    xin, hin);
            } else {
                gemv_gates<false><<<MTILES / 4, 128, 0, stream>>>(
                    gates,
                    nullptr, w_ih + (size_t)l * LAYER_ELEMS,
                    nullptr, w_hh + (size_t)l * LAYER_ELEMS,
                    xin, hin);
            }
            cell_update<<<H / 256, 256, 0, stream>>>(gates, hbuf + l * H, cbuf + l * H);
        }
        sample_step<<<1, 256, 0, stream>>>(hbuf + H, w_soft, w_emb, u, s, xv, xacc, out);
    }
}
